// ChunkDecoder_27479200759954
// MI455X (gfx1250) — matrix operand order repaired and checked by reading, compiled, not run on hardware
//
#include <hip/hip_runtime.h>
#include <hip/hip_bf16.h>
#include <math.h>

// ---------------------------------------------------------------------------
// DeBERTa disentangled attention for MI455X (gfx1250), wave32 + WMMA f16.
// B=16, N=512, DMODEL=768, H=12, HD=64, Z=512
// ---------------------------------------------------------------------------

typedef _Float16 v16h __attribute__((ext_vector_type(16)));
typedef float    v8f  __attribute__((ext_vector_type(8)));

#define BN      16
#define NN      512
#define DM      768
#define HH      12
#define HDD     64
#define ZZ      512

static __device__ __forceinline__ v8f wmma_f16(v16h a, v16h b, v8f c) {
  return __builtin_amdgcn_wmma_f32_16x16x32_f16(false, a, false, b, (short)0, c,
                                                false, false);
}

union frag_u { uint4 q[2]; v16h v; };

// A-matrix fragment (16x32 f16, MxK). LDS row-major [m][k], row stride ldm
// halves (multiple of 8). Two contiguous b128 loads per fragment.
static __device__ __forceinline__ v16h frag_a(const _Float16* lds, int m_loc,
                                              int ldm, int kbase, int lane) {
  int hiA = (lane & 16) ? 8 : 0;
  const uint4* p = (const uint4*)(lds + m_loc * ldm + kbase + hiA);
  frag_u u;
  u.q[0] = p[0];
  u.q[1] = p[2];  // +16 halves
  return u.v;
}

// B-matrix fragment (32x16 f16, KxN) with LDS stored [n][k]: two b128 loads.
static __device__ __forceinline__ v16h frag_b_nk(const _Float16* lds, int n_loc,
                                                 int ldn, int kbase, int lane) {
  int hiB = (lane & 16) ? 8 : 0;
  const uint4* p = (const uint4*)(lds + n_loc * ldn + kbase + hiB);
  frag_u u;
  u.q[0] = p[0];
  u.q[1] = p[2];  // +16 halves
  return u.v;
}

// CDNA5 async global->LDS copy (GLOBAL_LOAD_ASYNC_TO_LDS_B128, ASYNCcnt).
// lds_off = byte offset from the dynamic-LDS base (kernels have no static LDS).
template <int WIDTH, int ROWS, int NTHR>
static __device__ __forceinline__ void stage_async(const _Float16* __restrict__ src,
                                                   int src_stride_halves,
                                                   unsigned lds_off, int tid) {
  constexpr int VPR = WIDTH / 8;
  constexpr int ITERS = (ROWS * VPR) / NTHR;
#pragma unroll
  for (int e = 0; e < ITERS; ++e) {
    int v = tid + e * NTHR;
    int r = v / VPR, c = (v % VPR) * 8;
    const void* g = src + (long)r * src_stride_halves + c;
    unsigned l = lds_off + (unsigned)((r * WIDTH + c) * 2);
    asm volatile("global_load_async_to_lds_b128 %0, %1, off"
                 :: "v"(l), "v"(g) : "memory");
  }
}

static __device__ __forceinline__ void wait_async() {
  asm volatile("s_wait_asynccnt 0x0" ::: "memory");
}

// ---------------------------------------------------------------------------
// Kernel 0a: streaming f32 -> f16 convert (8 elements / thread).
// ---------------------------------------------------------------------------
__global__ __launch_bounds__(256) void cvt_kernel(const float* __restrict__ x,
                                                  _Float16* __restrict__ y,
                                                  long n) {
  long i = ((long)blockIdx.x * blockDim.x + threadIdx.x) * 8;
  if (i + 8 <= n) {
    float4 a = *(const float4*)(x + i);
    float4 b = *(const float4*)(x + i + 4);
    union { _Float16 h[8]; uint4 q; } u;
    u.h[0] = (_Float16)a.x; u.h[1] = (_Float16)a.y;
    u.h[2] = (_Float16)a.z; u.h[3] = (_Float16)a.w;
    u.h[4] = (_Float16)b.x; u.h[5] = (_Float16)b.y;
    u.h[6] = (_Float16)b.z; u.h[7] = (_Float16)b.w;
    *(uint4*)(y + i) = u.q;
  }
}

// ---------------------------------------------------------------------------
// Kernel 0b: Wt[n][k] = f16(W[k][n]), 64x64 tiles via padded LDS.
// ---------------------------------------------------------------------------
__global__ __launch_bounds__(256) void wt_kernel(const float* __restrict__ W,
                                                 _Float16* __restrict__ Wt) {
  __shared__ _Float16 lt[64][72];  // 72-half row stride: 16B aligned + skewed
  const int tid = threadIdx.x;
  const int k0 = blockIdx.x * 64, n0 = blockIdx.y * 64;
#pragma unroll
  for (int e = 0; e < 16; ++e) {
    int idx = tid + e * 256;
    int r = idx >> 6, c = idx & 63;  // r = k row, c = n col (coalesced)
    lt[c][r] = (_Float16)W[(long)(k0 + r) * DM + n0 + c];
  }
  __syncthreads();
#pragma unroll
  for (int e = 0; e < 2; ++e) {  // 64 rows x 8 vec16B
    int v = tid + e * 256;
    int r = v >> 3, c = (v & 7) * 8;  // r = n row, c = k col base
    *(uint4*)(Wt + (long)(n0 + r) * DM + k0 + c) = *(const uint4*)&lt[r][c];
  }
}

// ---------------------------------------------------------------------------
// Kernel 1: out = f16(A_h[M x 768] @ Wt_h^T + bias)   (all-f16)
// Block 128 threads / 4 waves; block tile 64x64; wave tile 16x64.
// Double-buffered async-to-LDS pipeline: tile k+1 streams in (ASYNCcnt) while
// tile k's WMMAs run. Dynamic LDS 16KB: A0@0 B0@4K A1@8K B1@12K.
// mode 0: out[i*768 + n]; mode 1 (values): out[((b*12+h)*64+d)*512 + j].
// ---------------------------------------------------------------------------
__global__ __launch_bounds__(128) void proj_kernel(
    const _Float16* __restrict__ A, const _Float16* __restrict__ Wt,
    const float* __restrict__ bias, _Float16* __restrict__ out, int mode) {
  extern __shared__ char psm[];
  const int tid = threadIdx.x, lane = tid & 31, w = tid >> 5;
  const int m0 = blockIdx.x * 64, n0 = blockIdx.y * 64;
  const int ml = lane & 15;

  v8f acc[4];
#pragma unroll
  for (int t = 0; t < 4; ++t) acc[t] = {};

  // prologue: issue tile 0
  stage_async<32, 64, 128>(A + (long)m0 * DM, DM, 0u, tid);
  stage_async<32, 64, 128>(Wt + (long)n0 * DM, DM, 4096u, tid);

  for (int k0 = 0; k0 < DM; k0 += 32) {
    const int p = (k0 >> 5) & 1;
    wait_async();      // tile k resident in LDS (per-wave)
    __syncthreads();   // all waves: tile k visible; tile k-1 reads retired
    if (k0 + 32 < DM) {
      unsigned nb = p ? 0u : 8192u;
      stage_async<32, 64, 128>(A + (long)m0 * DM + k0 + 32, DM, nb, tid);
      stage_async<32, 64, 128>(Wt + (long)n0 * DM + k0 + 32, DM, nb + 4096u, tid);
    }
    const _Float16* lA = (const _Float16*)(psm + (p ? 8192 : 0));
    const _Float16* lB = (const _Float16*)(psm + (p ? 12288 : 4096));
    v16h a = frag_a(lA, w * 16 + ml, 32, 0, lane);
#pragma unroll
    for (int t = 0; t < 4; ++t) {
      v16h b = frag_b_nk(lB, t * 16 + ml, 32, 0, lane);
      acc[t] = wmma_f16(a, b, acc[t]);
    }
  }

  const int hiC = (lane & 16) ? 8 : 0;
#pragma unroll
  for (int t = 0; t < 4; ++t) {
    int n = n0 + t * 16 + ml;
    float bv = bias[n];
#pragma unroll
    for (int r = 0; r < 8; ++r) {
      int i = m0 + w * 16 + r + hiC;
      _Float16 hv = (_Float16)(acc[t][r] + bv);
      if (mode == 0) {
        out[(long)i * DM + n] = hv;
      } else {
        int h = n >> 6, d = n & 63, b = i >> 9, j = i & 511;
        out[((long)((b * HH + h) * HDD + d)) * NN + j] = hv;
      }
    }
  }
}

// ---------------------------------------------------------------------------
// Kernel 2: kp[b][h][z][j] = pq[z,h,:] . ck[b,j,h,:]   (raw dot, f32)
// Dynamic LDS 16KB: lpq@0, lck@8K; async staging.
// ---------------------------------------------------------------------------
__global__ __launch_bounds__(128) void kp_kernel(
    const _Float16* __restrict__ pq_h, const _Float16* __restrict__ ck_h,
    float* __restrict__ kp) {
  extern __shared__ char ksm[];
  const _Float16* lpq = (const _Float16*)ksm;
  const _Float16* lck = (const _Float16*)(ksm + 8192);
  const int tid = threadIdx.x, lane = tid & 31, w = tid >> 5;
  const int tz = blockIdx.x >> 3, tj = blockIdx.x & 7;
  const int h = blockIdx.y, b = blockIdx.z;
  const int z0 = tz * 64, j0 = tj * 64;
  const int ml = lane & 15;

  stage_async<64, 64, 128>(pq_h + (long)z0 * DM + h * HDD, DM, 0u, tid);
  stage_async<64, 64, 128>(ck_h + ((long)(b * NN + j0)) * DM + h * HDD, DM, 8192u, tid);
  wait_async();
  __syncthreads();

  v8f acc[4];
#pragma unroll
  for (int t = 0; t < 4; ++t) acc[t] = {};
#pragma unroll
  for (int ks = 0; ks < 2; ++ks) {
    v16h a = frag_a(lpq, w * 16 + ml, 64, ks * 32, lane);
#pragma unroll
    for (int t = 0; t < 4; ++t) {
      v16h b2 = frag_b_nk(lck, t * 16 + ml, 64, ks * 32, lane);
      acc[t] = wmma_f16(a, b2, acc[t]);
    }
  }

  const int hiC = (lane & 16) ? 8 : 0;
  const long base = ((long)(b * HH + h)) * ZZ * NN;
#pragma unroll
  for (int t = 0; t < 4; ++t) {
    int j = j0 + t * 16 + ml;
#pragma unroll
    for (int r = 0; r < 8; ++r) {
      int z = z0 + w * 16 + r + hiC;
      kp[base + (long)z * NN + j] = acc[t][r];
    }
  }
}

// ---------------------------------------------------------------------------
// Kernel 3: fused attention per (b, h, 64-query tile). 256 threads / 8 waves.
// Dynamic LDS (no static LDS => offsets below are literal LDS offsets):
//   [0,      8K)  lq    cq tile 64x64 f16
//   [8K,    72K)  bufA  pk[512][64] -> ck[512][64] -> cvT[64][512] f16
//   [72K,  136K)  bufB  QP[64][512] -> exp-weights[64][512] f16
//   [136K, 140.5K) reduction scratch f32
// ---------------------------------------------------------------------------
#define LQ_OFF   0u
#define BUFA_OFF 8192u
#define BUFB_OFF (8192u + 65536u)
#define RED_OFF  (8192u + 2u * 65536u)

__global__ __launch_bounds__(256) void attn_kernel(
    const _Float16* __restrict__ cq_h, const _Float16* __restrict__ ck_h,
    const _Float16* __restrict__ cvT_h, const _Float16* __restrict__ pk_h,
    const float* __restrict__ kp_g, const int* __restrict__ relpos,
    const unsigned char* __restrict__ mask, const float* __restrict__ head_bias,
    float* __restrict__ out_pooled, float* __restrict__ out_attn) {
  extern __shared__ char smem[];
  _Float16* lq   = (_Float16*)(smem + LQ_OFF);    // 64*64
  _Float16* bufA = (_Float16*)(smem + BUFA_OFF);  // 32K halves
  _Float16* bufB = (_Float16*)(smem + BUFB_OFF);  // 64*512
  float* redM  = (float*)(smem + RED_OFF);        // 64*8
  float* redM2 = redM + 512;                      // 64
  float* redS  = redM2 + 64;                      // 64*8
  float* redS2 = redS + 512;                      // 64

  const int tid = threadIdx.x, lane = tid & 31, w = tid >> 5;
  const int qt = blockIdx.x, h = blockIdx.y, b = blockIdx.z;
  const int i0 = qt * 64;
  const int ml = lane & 15;
  const int hiC = (lane & 16) ? 8 : 0;

  // async stage: cq tile + pk slab
  stage_async<64, 64, 256>(cq_h + ((long)(b * NN + i0)) * DM + h * HDD, DM, LQ_OFF, tid);
  stage_async<64, 512, 256>(pk_h + (long)h * HDD, DM, BUFA_OFF, tid);
  wait_async();
  __syncthreads();

  // ---- Phase 1: QP = cq @ pk^T. Wave w owns z-strip w*64. ----
  v8f acc[16];
#pragma unroll
  for (int t = 0; t < 16; ++t) acc[t] = {};
#pragma unroll
  for (int ks = 0; ks < 2; ++ks) {
    v16h a4[4];
#pragma unroll
    for (int it = 0; it < 4; ++it) a4[it] = frag_a(lq, it * 16 + ml, 64, ks * 32, lane);
#pragma unroll
    for (int zt = 0; zt < 4; ++zt) {
      v16h bb = frag_b_nk(bufA, w * 64 + zt * 16 + ml, 64, ks * 32, lane);
#pragma unroll
      for (int it = 0; it < 4; ++it)
        acc[it * 4 + zt] = wmma_f16(a4[it], bb, acc[it * 4 + zt]);
    }
  }
  __syncthreads();  // all waves done reading pk from bufA

  // issue ck slab async load; it overlaps the QP writeback below
  stage_async<64, 512, 256>(ck_h + ((long)(b * NN)) * DM + h * HDD, DM, BUFA_OFF, tid);

  // QP -> bufB (f16)
#pragma unroll
  for (int zt = 0; zt < 4; ++zt) {
    int z = w * 64 + zt * 16 + ml;
#pragma unroll
    for (int it = 0; it < 4; ++it)
#pragma unroll
      for (int r = 0; r < 8; ++r)
        bufB[(it * 16 + r + hiC) * NN + z] = (_Float16)acc[it * 4 + zt][r];
  }
  wait_async();
  __syncthreads();  // ck + QP visible to all waves

  // ---- Phase 2: c2c accumulation (wave owns j-strip w*64). ----
#pragma unroll
  for (int t = 0; t < 16; ++t) acc[t] = {};
#pragma unroll
  for (int ks = 0; ks < 2; ++ks) {
    v16h a4[4];
#pragma unroll
    for (int it = 0; it < 4; ++it) a4[it] = frag_a(lq, it * 16 + ml, 64, ks * 32, lane);
#pragma unroll
    for (int jt = 0; jt < 4; ++jt) {
      v16h bb = frag_b_nk(bufA, w * 64 + jt * 16 + ml, 64, ks * 32, lane);
#pragma unroll
      for (int it = 0; it < 4; ++it)
        acc[it * 4 + jt] = wmma_f16(a4[it], bb, acc[it * 4 + jt]);
    }
  }
  __syncthreads();

  // ck dead; async-stage transposed cv slab [d][j] (contiguous in cvT_h).
  // Deliberately NOT waited here: the transfer overlaps phase 3 + softmax.
  stage_async<512, 64, 256>(cvT_h + ((long)(b * HH + h)) * HDD * NN, NN, BUFA_OFF, tid);

  // ---- Phase 3: combine c2c + c2p (LDS gather) + p2c (L2 gather). ----
  const float hb = head_bias[h];
  const float comb = 0.125f * 0.57735026919f;  // (1/sqrt(64)) * (1/sqrt(3))
  const long kpb = ((long)(b * HH + h)) * ZZ * NN;
#pragma unroll
  for (int it = 0; it < 4; ++it) {
#pragma unroll
    for (int jt = 0; jt < 4; ++jt) {
      int j = w * 64 + jt * 16 + ml;
      bool mb = mask[b * NN + j] != 0;
#pragma unroll
      for (int r = 0; r < 8; ++r) {
        int i_loc = it * 16 + r + hiC;
        int i = i0 + i_loc;
        int idx = relpos[((long)(b * NN + i)) * NN + j] + (ZZ / 2);
        idx = idx < 0 ? 0 : (idx > ZZ - 1 ? ZZ - 1 : idx);
        float cp = (float)bufB[i_loc * NN + idx];
        float pc = kp_g[kpb + (long)idx * NN + j];
        float raw = (acc[it * 4 + jt][r] + cp + pc) * comb + hb;
        out_attn[(((long)(b * NN + i)) * NN + j) * HH + h] = raw;
        acc[it * 4 + jt][r] = mb ? raw : -1000000.0f;
      }
    }
  }
  __syncthreads();  // all QP reads done; bufB reusable for exp-weights

  // ---- Softmax over j ----
#pragma unroll
  for (int it = 0; it < 4; ++it) {
#pragma unroll
    for (int r = 0; r < 8; ++r) {
      float m = -3.0e38f;
#pragma unroll
      for (int jt = 0; jt < 4; ++jt) m = fmaxf(m, acc[it * 4 + jt][r]);
#pragma unroll
      for (int d = 1; d <= 8; d <<= 1) m = fmaxf(m, __shfl_xor(m, d, 32));
      if (ml == 0) redM[(it * 16 + r + hiC) * 8 + w] = m;
    }
  }
  __syncthreads();
  if (tid < 64) {
    float m = redM[tid * 8];
#pragma unroll
    for (int k = 1; k < 8; ++k) m = fmaxf(m, redM[tid * 8 + k]);
    redM2[tid] = m;
  }
  __syncthreads();

#pragma unroll
  for (int it = 0; it < 4; ++it) {
#pragma unroll
    for (int r = 0; r < 8; ++r) {
      int i_loc = it * 16 + r + hiC;
      float rm = redM2[i_loc];
      float s = 0.0f;
#pragma unroll
      for (int jt = 0; jt < 4; ++jt) {
        float e = __expf(acc[it * 4 + jt][r] - rm);
        bufB[i_loc * NN + (w * 64 + jt * 16 + ml)] = (_Float16)e;
        s += e;
      }
#pragma unroll
      for (int d = 1; d <= 8; d <<= 1) s += __shfl_xor(s, d, 32);
      if (ml == 0) redS[i_loc * 8 + w] = s;
    }
  }
  __syncthreads();
  if (tid < 64) {
    float s = 0.0f;
#pragma unroll
    for (int k = 0; k < 8; ++k) s += redS[tid * 8 + k];
    redS2[tid] = s;
  }
  wait_async();    // cv slab async loads complete (per-wave), then rendezvous
  __syncthreads();

  // ---- Phase 4: pooled = (Wexp @ CV) / rowsum ----
  const int ti = w >> 1;
  v8f pac[2];
#pragma unroll
  for (int t = 0; t < 2; ++t) pac[t] = {};
  for (int ks = 0; ks < 16; ++ks) {
    v16h a = frag_a(bufB, ti * 16 + ml, NN, ks * 32, lane);
#pragma unroll
    for (int t = 0; t < 2; ++t) {
      int nt = (w & 1) * 2 + t;
      v16h bb = frag_b_nk(bufA, nt * 16 + ml, NN, ks * 32, lane);
      pac[t] = wmma_f16(a, bb, pac[t]);
    }
  }
#pragma unroll
  for (int t = 0; t < 2; ++t) {
    int d = ((w & 1) * 2 + t) * 16 + ml;
#pragma unroll
    for (int r = 0; r < 8; ++r) {
      int i_loc = ti * 16 + r + hiC;
      float val = pac[t][r] / redS2[i_loc];
      out_pooled[((long)(b * NN + i0 + i_loc)) * DM + h * HDD + d] = val;
    }
  }
}

// ---------------------------------------------------------------------------
extern "C" void kernel_launch(void* const* d_in, const int* in_sizes, int n_in,
                              void* d_out, int out_size, void* d_ws,
                              size_t ws_size, hipStream_t stream) {
  (void)in_sizes; (void)n_in; (void)out_size; (void)ws_size;

  const float* cqueries  = (const float*)d_in[0];
  const float* ckeys     = (const float*)d_in[1];
  const float* cvalues   = (const float*)d_in[2];
  const unsigned char* mask = (const unsigned char*)d_in[3];
  const int*   relpos    = (const int*)d_in[4];
  const float* pos_emb   = (const float*)d_in[5];
  const float* wq = (const float*)d_in[6];
  const float* bq = (const float*)d_in[7];
  const float* wk = (const float*)d_in[8];
  const float* bk = (const float*)d_in[9];
  const float* wv = (const float*)d_in[10];
  const float* bv = (const float*)d_in[11];
  const float* pk_w = (const float*)d_in[12];
  const float* pk_b = (const float*)d_in[13];
  const float* pq_w = (const float*)d_in[14];
  const float* pq_b = (const float*)d_in[15];
  const float* head_bias = (const float*)d_in[16];

  const long ACT = (long)BN * NN * DM;  // 6291456
  const long PE  = (long)ZZ * DM;       // 393216
  const long WW  = (long)DM * DM;       // 589824

  // Workspace layout (f16 unless noted)
  _Float16* p = (_Float16*)d_ws;
  _Float16* xq_h = p; p += ACT;
  _Float16* xk_h = p; p += ACT;
  _Float16* xv_h = p; p += ACT;
  _Float16* pe_h = p; p += PE;
  _Float16* wtq  = p; p += WW;
  _Float16* wtk  = p; p += WW;
  _Float16* wtv  = p; p += WW;
  _Float16* wtpk = p; p += WW;
  _Float16* wtpq = p; p += WW;
  _Float16* cq_h = p; p += ACT;
  _Float16* ck_h = p; p += ACT;
  _Float16* cvT_h = p; p += ACT;   // [b][h][d][j]
  _Float16* pk_h = p; p += PE;
  _Float16* pq_h = p; p += PE;
  float* kp_g = (float*)p;         // 16*12*512*512 f32

  float* out_pooled = (float*)d_out;
  float* out_attn   = out_pooled + (long)BN * NN * DM;

  // 0a: activations/pos-emb -> f16
  cvt_kernel<<<dim3((unsigned)(ACT / 8 / 256)), dim3(256), 0, stream>>>(cqueries, xq_h, ACT);
  cvt_kernel<<<dim3((unsigned)(ACT / 8 / 256)), dim3(256), 0, stream>>>(ckeys,    xk_h, ACT);
  cvt_kernel<<<dim3((unsigned)(ACT / 8 / 256)), dim3(256), 0, stream>>>(cvalues,  xv_h, ACT);
  cvt_kernel<<<dim3((unsigned)(PE  / 8 / 256)), dim3(256), 0, stream>>>(pos_emb,  pe_h, PE);

  // 0b: weight transposes -> f16 [n][k]
  dim3 wtg(DM / 64, DM / 64);
  wt_kernel<<<wtg, dim3(256), 0, stream>>>(wq,   wtq);
  wt_kernel<<<wtg, dim3(256), 0, stream>>>(wk,   wtk);
  wt_kernel<<<wtg, dim3(256), 0, stream>>>(wv,   wtv);
  wt_kernel<<<wtg, dim3(256), 0, stream>>>(pk_w, wtpk);
  wt_kernel<<<wtg, dim3(256), 0, stream>>>(pq_w, wtpq);

  // 1: projections (double-buffered async pipeline, 16KB dynamic LDS)
  dim3 pb(128);
  proj_kernel<<<dim3((BN * NN) / 64, DM / 64), pb, 16384, stream>>>(xq_h, wtq, bq, cq_h, 0);
  proj_kernel<<<dim3((BN * NN) / 64, DM / 64), pb, 16384, stream>>>(xk_h, wtk, bk, ck_h, 0);
  proj_kernel<<<dim3((BN * NN) / 64, DM / 64), pb, 16384, stream>>>(xv_h, wtv, bv, cvT_h, 1);
  proj_kernel<<<dim3(ZZ / 64, DM / 64), pb, 16384, stream>>>(pe_h, wtpk, pk_b, pk_h, 0);
  proj_kernel<<<dim3(ZZ / 64, DM / 64), pb, 16384, stream>>>(pe_h, wtpq, pq_b, pq_h, 0);

  // 2: p2c source table
  kp_kernel<<<dim3((ZZ / 64) * (NN / 64), HH, BN), pb, 16384, stream>>>(pq_h, ck_h, kp_g);

  // 3: fused attention
  size_t lds_bytes = 8192 + 2 * 65536 + (512 + 64 + 512 + 64) * sizeof(float);
  attn_kernel<<<dim3(NN / 64, HH, BN), dim3(256), lds_bytes, stream>>>(
      cq_h, ck_h, cvT_h, pk_h, kp_g, relpos, mask, head_bias, out_pooled, out_attn);
}
